// Head_7258494730366
// MI455X (gfx1250) — compile-verified, matrix operand order checked by reading, not run on hardware
//
#include <hip/hip_runtime.h>
#include <hip/hip_bf16.h>
#include <math.h>

// Attention head: x[B,T,C] fp32, Wq/Wk/Wv[C,D] fp32 -> out[B,T,D] fp32.
// B=4 T=4096 C=1024 D=128.  bf16 WMMA (v_wmma_f32_16x16x32_bf16) everywhere,
// f32 accumulation.  Flash-attention kernel uses double-buffered
// GLOBAL_LOAD_ASYNC_TO_LDS_B128 copies (ASYNCcnt) to overlap tile movement
// with WMMA compute.
// Workspace: Q[B*T,D] | K[B*T,D] | Vt[B,D,T]  (bf16, 4 MB each, 12 MB).

typedef __bf16 bf16_t;
typedef __attribute__((ext_vector_type(16))) __bf16 v16bf;
typedef __attribute__((ext_vector_type(8)))  __bf16 v8bf;
typedef __attribute__((ext_vector_type(8)))  float  v8f;

#define BATCH 4
#define TLEN  4096
#define CDIM  1024
#define DDIM  128
#define NROWS (BATCH * TLEN)   // 16384

__device__ __forceinline__ bf16_t f2bf(float f) { return (bf16_t)f; }

__device__ __forceinline__ v16bf cat16(v8bf lo, v8bf hi) {
    return __builtin_shufflevector(lo, hi, 0,1,2,3,4,5,6,7,8,9,10,11,12,13,14,15);
}

__device__ __forceinline__ v8f wmma_bf16(v16bf a, v16bf b, v8f c) {
    return __builtin_amdgcn_wmma_f32_16x16x32_bf16(
        /*neg_a=*/false, a, /*neg_b=*/false, b,
        /*c_mod=*/(short)0, c, /*reuse_a=*/false, /*reuse_b=*/false);
}

// Async 16-byte global -> LDS copy (per-lane), tracked on ASYNCcnt.
// LDS offset = low 32 bits of the generic shared-memory pointer
// (flat-LDS addressing uses addr[31:0] as the LDS byte offset).
__device__ __forceinline__ void async_copy_b128(const void* gptr, void* lptr) {
    unsigned loff = (unsigned)(unsigned long long)lptr;
    asm volatile("global_load_async_to_lds_b128 %0, %1, off"
                 :: "v"(loff), "v"(gptr) : "memory");
}
__device__ __forceinline__ void wait_asynccnt0() {
    asm volatile("s_wait_asynccnt 0x0" ::: "memory");
}

// ---------------------------------------------------------------------------
// Kernel 1: QKV projection.  grid.x = NROWS/128, grid.y = 3 (q,k,v).
// 256 threads = 8 waves; wave computes a 16x128 strip.  K-loop steps 32.
// Q is pre-scaled by D^-1/2.  V is stored transposed [B][D][T] so the
// attention kernel can async-copy its V tiles without an LDS transpose.
// ---------------------------------------------------------------------------
__global__ __launch_bounds__(256)
void qkv_gemm_kernel(const float* __restrict__ x,
                     const float* __restrict__ Wq,
                     const float* __restrict__ Wk,
                     const float* __restrict__ Wv,
                     bf16_t* __restrict__ qb,
                     bf16_t* __restrict__ kb,
                     bf16_t* __restrict__ vb,
                     float qscale)
{
    __shared__ __align__(16) bf16_t xs [128][40];  // x tile  [row][k], bf16
    __shared__ __align__(16) bf16_t wst[128][40];  // W tile transposed [d][k]

    const int tid  = threadIdx.x;
    const int wv   = tid >> 5;
    const int lane = tid & 31;
    const int half = lane >> 4;
    const int l16  = lane & 15;
    const int row0 = blockIdx.x * 128;
    const int sel  = blockIdx.y;

    const float* W = (sel == 0) ? Wq : (sel == 1) ? Wk : Wv;
    bf16_t* out    = (sel == 0) ? qb : (sel == 1) ? kb : vb;
    const float sc = (sel == 0) ? qscale : 1.0f;

    v8f acc[8];
#pragma unroll
    for (int t = 0; t < 8; ++t)
#pragma unroll
        for (int i = 0; i < 8; ++i) acc[t][i] = 0.0f;

    for (int k0 = 0; k0 < CDIM; k0 += 32) {
        // x tile: 128 rows x 32 k (fp32 -> bf16).  1024 float4 slots.
#pragma unroll
        for (int i = 0; i < 4; ++i) {
            int s  = tid + i * 256;
            int r  = s >> 3;
            int c4 = (s & 7) * 4;
            const float* gp = x + (size_t)(row0 + r) * CDIM + k0 + c4;
            float4 v = *(const float4*)gp;
            if (k0 + 32 < CDIM) __builtin_prefetch(gp + 32, 0, 1);
            xs[r][c4 + 0] = f2bf(v.x);
            xs[r][c4 + 1] = f2bf(v.y);
            xs[r][c4 + 2] = f2bf(v.z);
            xs[r][c4 + 3] = f2bf(v.w);
        }
        // W tile: 32 k x 128 d, stored transposed.  1024 float4 slots.
#pragma unroll
        for (int i = 0; i < 4; ++i) {
            int s  = tid + i * 256;
            int r  = s >> 5;           // k row (0..31)
            int c4 = (s & 31) * 4;     // d col
            float4 v = *(const float4*)(W + (size_t)(k0 + r) * DDIM + c4);
            wst[c4 + 0][r] = f2bf(v.x);
            wst[c4 + 1][r] = f2bf(v.y);
            wst[c4 + 2][r] = f2bf(v.z);
            wst[c4 + 3][r] = f2bf(v.w);
        }
        __syncthreads();

        // A fragment (16x32): lane covers K in [half*8,+8) and [16+half*8,+8)
        const bf16_t* xrow = &xs[wv * 16 + l16][0];
        v16bf afrag = cat16(*(const v8bf*)(xrow + half * 8),
                            *(const v8bf*)(xrow + 16 + half * 8));
#pragma unroll
        for (int nt = 0; nt < 8; ++nt) {
            const bf16_t* wrow = &wst[nt * 16 + l16][0];
            v16bf bfrag = cat16(*(const v8bf*)(wrow + half * 8),
                                *(const v8bf*)(wrow + 16 + half * 8));
            acc[nt] = wmma_bf16(afrag, bfrag, acc[nt]);
        }
        __syncthreads();
    }

    // Store.  C-layout: VGPR r, lane -> (m = r + 8*half, n = l16).
    if (sel == 2) {
        // V: transposed layout [B][D][T]
#pragma unroll
        for (int t = 0; t < 8; ++t)
#pragma unroll
            for (int r = 0; r < 8; ++r) {
                int m    = r + half * 8;
                int grow = row0 + wv * 16 + m;      // == b*T + tpos
                int bb   = grow >> 12;              // / TLEN
                int tp   = grow & (TLEN - 1);
                int n    = t * 16 + l16;
                out[((size_t)(bb * DDIM + n)) * TLEN + tp] = f2bf(acc[t][r]);
            }
    } else {
#pragma unroll
        for (int t = 0; t < 8; ++t)
#pragma unroll
            for (int r = 0; r < 8; ++r) {
                int m = r + half * 8;
                size_t idx = (size_t)(row0 + wv * 16 + m) * DDIM + t * 16 + l16;
                out[idx] = f2bf(acc[t][r] * sc);
            }
    }
}

// ---------------------------------------------------------------------------
// Kernel 2: causal flash attention.  grid.x = T/128, grid.y = B; 8 waves.
// Wave owns 16 query rows; streams key blocks of 32.  K/V tiles arrive via
// double-buffered async global->LDS copies overlapped with WMMA compute.
// ---------------------------------------------------------------------------
__global__ __launch_bounds__(256)
void flash_attn_kernel(const bf16_t* __restrict__ Q,
                       const bf16_t* __restrict__ K,
                       const bf16_t* __restrict__ Vt,
                       float* __restrict__ out)
{
    __shared__ __align__(16) bf16_t kt[2][32][136];   // K tile [key][d]
    __shared__ __align__(16) bf16_t vt[2][128][40];   // V tile [d][key]
    __shared__ __align__(16) bf16_t ps[8][16][40];    // per-wave P scratch

    const int tid  = threadIdx.x;
    const int wv   = tid >> 5;
    const int lane = tid & 31;
    const int half = lane >> 4;
    const int l16  = lane & 15;
    const int b    = blockIdx.y;
    const int q0   = blockIdx.x * 128;

    // Per-thread async-copy slots (2 x 16B for K, 2 x 16B for V per tile).
    const int kr  = tid >> 4;              // K: key row 0..15 (+16 on 2nd)
    const int kc8 = (tid & 15) * 8;        // K: d offset
    const int vr  = tid >> 2;              // V: d row 0..63 (+64 on 2nd)
    const int vc  = (tid & 3) * 8;         // V: key offset

    // Issue async copies for key-block j into buffer p.
    auto issue_tile = [&](int j, int p) {
#pragma unroll
        for (int i = 0; i < 2; ++i) {
            int r = kr + i * 16;
            async_copy_b128(K + ((size_t)b * TLEN + j + r) * DDIM + kc8,
                            &kt[p][r][kc8]);
        }
#pragma unroll
        for (int i = 0; i < 2; ++i) {
            int r = vr + i * 64;
            async_copy_b128(Vt + ((size_t)b * DDIM + r) * TLEN + j + vc,
                            &vt[p][r][vc]);
        }
    };

    // Load this lane's Q A-fragments for all 4 K=32 chunks of D=128.
    const bf16_t* qrow = Q + ((size_t)b * TLEN + q0 + wv * 16 + l16) * DDIM;
    v16bf qf[4];
#pragma unroll
    for (int c = 0; c < 4; ++c)
        qf[c] = cat16(*(const v8bf*)(qrow + c * 32 + half * 8),
                      *(const v8bf*)(qrow + c * 32 + 16 + half * 8));

    v8f o[8];
    float m_run[8], l_run[8];
#pragma unroll
    for (int t = 0; t < 8; ++t)
#pragma unroll
        for (int i = 0; i < 8; ++i) o[t][i] = 0.0f;
#pragma unroll
    for (int r = 0; r < 8; ++r) { m_run[r] = -INFINITY; l_run[r] = 0.0f; }

    const int kend = q0 + 128;             // causal: keys < q0+128 needed
    issue_tile(0, 0);

    int p = 0;
    for (int j = 0; j < kend; j += 32, p ^= 1) {
        wait_asynccnt0();
        __syncthreads();                   // tile j visible to all waves;
                                           // also: all waves done with buf p^1
        if (j + 32 < kend) issue_tile(j + 32, p ^ 1);

        // S = Q * K^T  (two 16-key N tiles; contract D via 4 WMMAs each)
        v8f s0, s1;
#pragma unroll
        for (int i = 0; i < 8; ++i) { s0[i] = 0.0f; s1[i] = 0.0f; }
#pragma unroll
        for (int c = 0; c < 4; ++c) {
            const bf16_t* kp0 = &kt[p][l16][c * 32];
            v16bf b0 = cat16(*(const v8bf*)(kp0 + half * 8),
                             *(const v8bf*)(kp0 + 16 + half * 8));
            s0 = wmma_bf16(qf[c], b0, s0);
            const bf16_t* kp1 = &kt[p][16 + l16][c * 32];
            v16bf b1 = cat16(*(const v8bf*)(kp1 + half * 8),
                             *(const v8bf*)(kp1 + 16 + half * 8));
            s1 = wmma_bf16(qf[c], b1, s1);
        }

        // Causal mask + online softmax (rows live across 16-lane halves).
#pragma unroll
        for (int r = 0; r < 8; ++r) {
            const int qg  = q0 + wv * 16 + r + half * 8;   // global query row
            const int kg0 = j + l16;
            const int kg1 = j + 16 + l16;
            float v0 = (kg0 > qg) ? -INFINITY : s0[r];
            float v1 = (kg1 > qg) ? -INFINITY : s1[r];

            float rm = fmaxf(v0, v1);
            rm = fmaxf(rm, __shfl_xor(rm, 1));
            rm = fmaxf(rm, __shfl_xor(rm, 2));
            rm = fmaxf(rm, __shfl_xor(rm, 4));
            rm = fmaxf(rm, __shfl_xor(rm, 8));

            float mnew  = fmaxf(m_run[r], rm);
            float alpha = __expf(m_run[r] - mnew);
            float p0 = __expf(v0 - mnew);
            float p1 = __expf(v1 - mnew);
            float rs = p0 + p1;
            rs += __shfl_xor(rs, 1);
            rs += __shfl_xor(rs, 2);
            rs += __shfl_xor(rs, 4);
            rs += __shfl_xor(rs, 8);

            l_run[r] = l_run[r] * alpha + rs;
            m_run[r] = mnew;
#pragma unroll
            for (int t = 0; t < 8; ++t) o[t][r] *= alpha;

            const int m = r + half * 8;
            ps[wv][m][l16]      = f2bf(p0);
            ps[wv][m][16 + l16] = f2bf(p1);
        }

        // O += P * V   (per-wave LDS scratch; same-wave DS ordering suffices)
        const bf16_t* prow = &ps[wv][l16][0];
        v16bf pf = cat16(*(const v8bf*)(prow + half * 8),
                         *(const v8bf*)(prow + 16 + half * 8));
#pragma unroll
        for (int t = 0; t < 8; ++t) {
            const bf16_t* vrow = &vt[p][t * 16 + l16][0];
            v16bf bv = cat16(*(const v8bf*)(vrow + half * 8),
                             *(const v8bf*)(vrow + 16 + half * 8));
            o[t] = wmma_bf16(pf, bv, o[t]);
        }
        // No trailing barrier: the leading barrier of iteration j+32 orders
        // all waves' reads of buf p before that iteration's copy of tile j+64 overwrites it.
    }

    // Normalize and store fp32 output.
#pragma unroll
    for (int r = 0; r < 8; ++r) {
        const int m = r + half * 8;
        const float inv = 1.0f / l_run[r];
        const size_t rowi = ((size_t)b * TLEN + q0 + wv * 16 + m) * DDIM;
#pragma unroll
        for (int t = 0; t < 8; ++t)
            out[rowi + t * 16 + l16] = o[t][r] * inv;
    }
}

// ---------------------------------------------------------------------------
extern "C" void kernel_launch(void* const* d_in, const int* in_sizes, int n_in,
                              void* d_out, int out_size, void* d_ws, size_t ws_size,
                              hipStream_t stream) {
    (void)in_sizes; (void)n_in; (void)out_size; (void)ws_size;
    const float* x  = (const float*)d_in[0];
    const float* Wq = (const float*)d_in[1];
    const float* Wk = (const float*)d_in[2];
    const float* Wv = (const float*)d_in[3];
    float* out = (float*)d_out;

    bf16_t* qb = (bf16_t*)d_ws;                       // 4 MB  [B*T, D]
    bf16_t* kb = qb + (size_t)NROWS * DDIM;           // 4 MB  [B*T, D]
    bf16_t* vb = kb + (size_t)NROWS * DDIM;           // 4 MB  [B, D, T]

    const float qscale = 1.0f / sqrtf((float)DDIM);   // folded into Q

    dim3 g1(NROWS / 128, 3);
    qkv_gemm_kernel<<<g1, 256, 0, stream>>>(x, Wq, Wk, Wv, qb, kb, vb, qscale);

    dim3 g2(TLEN / 128, BATCH);
    flash_attn_kernel<<<g2, 256, 0, stream>>>(qb, kb, vb, out);
}
